// CTI_toC_4956392259959
// MI455X (gfx1250) — matrix operand order repaired and checked by reading, not compiled, not run on hardware
//
#include <hip/hip_runtime.h>
#include <math.h>

// ---------------- CDNA5 vector types ----------------
typedef __attribute__((ext_vector_type(16))) __bf16 v16bf;
typedef __attribute__((ext_vector_type(8)))  __bf16 v8bf;
typedef __attribute__((ext_vector_type(8)))  float  v8f;
typedef __attribute__((ext_vector_type(4)))  unsigned int u32x4;
typedef __attribute__((ext_vector_type(8)))  int i32x8;
typedef __attribute__((ext_vector_type(4)))  int i32x4;

#define DIMC   384
#define NTOK   5376
#define NHEAD  6
#define HDIM   64
#define LN_EPS 1e-6f

// --- Tensor Data Mover availability / arity discrimination ---
#if defined(__has_builtin)
#if __has_builtin(__builtin_amdgcn_tensor_load_to_lds) && __has_builtin(__builtin_amdgcn_s_wait_tensorcnt)
#define CTI_USE_TDM 1
#endif
#endif
#if defined(__has_include)
#if __has_include(<hip/amd_detail/amd_gfx1250_TDM.h>)
#define CTI_TDM_6ARG 1
#endif
#endif

#ifdef CTI_USE_TDM
#pragma message("CTI: TDM builtin path ENABLED (tensor_load_to_lds)")
#else
#pragma message("CTI: TDM builtin ABSENT - cooperative-load fallback")
#endif
#ifdef CTI_TDM_6ARG
#pragma message("CTI: 6-arg TDM toolchain (therock headers present)")
#else
#pragma message("CTI: 5-arg TDM toolchain assumed (no therock TDM header)")
#endif

// ---------------- wave32 reduction ----------------
__device__ __forceinline__ float wred(float x) {
    x += __shfl_xor(x, 16, 32);
    x += __shfl_xor(x,  8, 32);
    x += __shfl_xor(x,  4, 32);
    x += __shfl_xor(x,  2, 32);
    x += __shfl_xor(x,  1, 32);
    return x;
}

// =====================================================================
// Kernel 1: build q (+feat on middle segment) + nested LayerNorm chain.
// One wave per token row (12 f32 per lane).
// =====================================================================
__global__ __launch_bounds__(256) void k_ln_fuse(
    const float* __restrict__ query, const float* __restrict__ feat,
    const float* __restrict__ qn_g, const float* __restrict__ qn_b,
    const float* __restrict__ fn_g, const float* __restrict__ fn_b,
    const float* __restrict__ iqn_g, const float* __restrict__ iqn_b,
    const float* __restrict__ ifn_g, const float* __restrict__ ifn_b,
    float* __restrict__ qn, float* __restrict__ iqn, float* __restrict__ ifn)
{
    const int wid  = threadIdx.x >> 5;
    const int lane = threadIdx.x & 31;
    const int row  = blockIdx.x * 8 + wid;
    if (row >= NTOK) return;

    float x[12];
    const float* qp = query + (size_t)row * DIMC;
#pragma unroll
    for (int j = 0; j < 3; ++j) {
        float4 v = *(const float4*)(qp + lane * 4 + 128 * j);
        x[j*4+0] = v.x; x[j*4+1] = v.y; x[j*4+2] = v.z; x[j*4+3] = v.w;
    }
    if (row >= 4096 && row < 5120) {
        const float* fp = feat + (size_t)(row - 4096) * DIMC;
#pragma unroll
        for (int j = 0; j < 3; ++j) {
            float4 v = *(const float4*)(fp + lane * 4 + 128 * j);
            x[j*4+0] += v.x; x[j*4+1] += v.y; x[j*4+2] += v.z; x[j*4+3] += v.w;
        }
    }

    float s = 0.f, ss = 0.f;
#pragma unroll
    for (int j = 0; j < 12; ++j) { s += x[j]; ss += x[j] * x[j]; }
    s = wred(s); ss = wred(ss);
    const float m = s * (1.f / DIMC);
    const float r = rsqrtf(ss * (1.f / DIMC) - m * m + LN_EPS);

    float qv[12], fv[12];
#pragma unroll
    for (int j = 0; j < 3; ++j) {
        const int idx = lane * 4 + 128 * j;
        float4 gq = *(const float4*)(qn_g + idx), bq = *(const float4*)(qn_b + idx);
        float4 gf = *(const float4*)(fn_g + idx), bf = *(const float4*)(fn_b + idx);
        float n0 = (x[j*4+0] - m) * r, n1 = (x[j*4+1] - m) * r;
        float n2 = (x[j*4+2] - m) * r, n3 = (x[j*4+3] - m) * r;
        qv[j*4+0] = n0 * gq.x + bq.x; qv[j*4+1] = n1 * gq.y + bq.y;
        qv[j*4+2] = n2 * gq.z + bq.z; qv[j*4+3] = n3 * gq.w + bq.w;
        fv[j*4+0] = n0 * gf.x + bf.x; fv[j*4+1] = n1 * gf.y + bf.y;
        fv[j*4+2] = n2 * gf.z + bf.z; fv[j*4+3] = n3 * gf.w + bf.w;
    }

    float sq = 0.f, ssq = 0.f, sf = 0.f, ssf = 0.f;
#pragma unroll
    for (int j = 0; j < 12; ++j) {
        sq += qv[j]; ssq += qv[j] * qv[j];
        sf += fv[j]; ssf += fv[j] * fv[j];
    }
    sq = wred(sq); ssq = wred(ssq); sf = wred(sf); ssf = wred(ssf);
    const float mq = sq * (1.f / DIMC);
    const float rq = rsqrtf(ssq * (1.f / DIMC) - mq * mq + LN_EPS);
    const float mf = sf * (1.f / DIMC);
    const float rf = rsqrtf(ssf * (1.f / DIMC) - mf * mf + LN_EPS);

    float* qnp  = qn  + (size_t)row * DIMC;
    float* iqnp = iqn + (size_t)row * DIMC;
    float* ifnp = ifn + (size_t)row * DIMC;
#pragma unroll
    for (int j = 0; j < 3; ++j) {
        const int idx = lane * 4 + 128 * j;
        float4 giq = *(const float4*)(iqn_g + idx), biq = *(const float4*)(iqn_b + idx);
        float4 gif = *(const float4*)(ifn_g + idx), bif = *(const float4*)(ifn_b + idx);
        float4 o;
        o.x = qv[j*4+0]; o.y = qv[j*4+1]; o.z = qv[j*4+2]; o.w = qv[j*4+3];
        *(float4*)(qnp + idx) = o;
        o.x = (qv[j*4+0]-mq)*rq*giq.x + biq.x; o.y = (qv[j*4+1]-mq)*rq*giq.y + biq.y;
        o.z = (qv[j*4+2]-mq)*rq*giq.z + biq.z; o.w = (qv[j*4+3]-mq)*rq*giq.w + biq.w;
        *(float4*)(iqnp + idx) = o;
        o.x = (fv[j*4+0]-mf)*rf*gif.x + bif.x; o.y = (fv[j*4+1]-mf)*rf*gif.y + bif.y;
        o.z = (fv[j*4+2]-mf)*rf*gif.z + bif.z; o.w = (fv[j*4+3]-mf)*rf*gif.w + bif.w;
        *(float4*)(ifnp + idx) = o;
    }
}

// =====================================================================
// Kernel 2: single LayerNorm (for FFN input)
// =====================================================================
__global__ __launch_bounds__(256) void k_ln(
    const float* __restrict__ in, const float* __restrict__ g,
    const float* __restrict__ b, float* __restrict__ out)
{
    const int wid  = threadIdx.x >> 5;
    const int lane = threadIdx.x & 31;
    const int row  = blockIdx.x * 8 + wid;
    if (row >= NTOK) return;

    float x[12];
    const float* ip = in + (size_t)row * DIMC;
#pragma unroll
    for (int j = 0; j < 3; ++j) {
        float4 v = *(const float4*)(ip + lane * 4 + 128 * j);
        x[j*4+0]=v.x; x[j*4+1]=v.y; x[j*4+2]=v.z; x[j*4+3]=v.w;
    }
    float s = 0.f, ss = 0.f;
#pragma unroll
    for (int j = 0; j < 12; ++j) { s += x[j]; ss += x[j]*x[j]; }
    s = wred(s); ss = wred(ss);
    const float m = s * (1.f / DIMC);
    const float r = rsqrtf(ss * (1.f / DIMC) - m * m + LN_EPS);
    float* op = out + (size_t)row * DIMC;
#pragma unroll
    for (int j = 0; j < 3; ++j) {
        const int idx = lane * 4 + 128 * j;
        float4 gg = *(const float4*)(g + idx), bb = *(const float4*)(b + idx);
        float4 o;
        o.x = (x[j*4+0]-m)*r*gg.x + bb.x; o.y = (x[j*4+1]-m)*r*gg.y + bb.y;
        o.z = (x[j*4+2]-m)*r*gg.z + bb.z; o.w = (x[j*4+3]-m)*r*gg.w + bb.w;
        *(float4*)(op + idx) = o;
    }
}

// =====================================================================
// Kernel 3: WMMA GEMM,  C[M,N] = A[M,K] @ W[K,N] (+bias) (+R residual)
// Block tile 64x64, 8 waves (4Mx2N); each wave: one A fragment reused
// over two 16x16 N-subtiles -> 6x v_wmma_f32_16x16x32_bf16 per K-step.
// A tile (64x32 f32) is double-buffered via the Tensor Data Mover
// (tensor_load_to_lds, TENSORcnt, one tile in flight); W tile staged
// with transpose + fp32->bf16 hi/lo split.  K % 32 == 0.
// =====================================================================
__global__ __launch_bounds__(256) void k_gemm(
    const float* __restrict__ A, const float* __restrict__ W,
    const float* __restrict__ bias, const float* __restrict__ R,
    float* __restrict__ C, int M, int N, int K)
{
    __shared__ __align__(16) float  Araw[2][64][32];  // raw f32 A tiles (TDM dest)
    __shared__ __align__(32) __bf16 Bhi[64][32];      // transposed: [n][k]
    __shared__ __align__(32) __bf16 Blo[64][32];

    const int tid  = threadIdx.x;
    const int wid  = tid >> 5;
    const int lane = tid & 31;
    const int wm   = wid >> 1;          // 0..3 (M sub-tile)
    const int wn   = wid & 1;           // 0..1 (N sub-tile pair)
    const int m0   = blockIdx.x * 64;
    const int n0   = blockIdx.y * 64;

    v8f acc0 = {0.f, 0.f, 0.f, 0.f, 0.f, 0.f, 0.f, 0.f};
    v8f acc1 = {0.f, 0.f, 0.f, 0.f, 0.f, 0.f, 0.f, 0.f};

    const int bk = tid >> 3;            // 0..31  W tile k-row
    const int bn = (tid & 7) * 8;       // 0..56

    const int nsteps = K >> 5;

#ifdef CTI_USE_TDM
    auto tdm_issue = [&](int buf, int k0) {
        const unsigned long long ga =
            (unsigned long long)(A + (size_t)m0 * K + k0);
        u32x4 g0;
        g0[0] = 1u;                                   // count=1, user desc
        g0[1] = (unsigned)(size_t)(&Araw[buf][0][0]); // lds_addr
        g0[2] = (unsigned)(ga & 0xFFFFFFFFull);       // global_addr[31:0]
        g0[3] = (unsigned)(ga >> 32) | (2u << 30);    // addr[56:32] | type=2
        i32x8 g1;
        g1[0] = 2 << 16;                              // data_size = 4B
        g1[1] = (K & 0xFFFF) << 16;                   // tensor_dim0 lo16
        g1[2] = ((M & 0xFFFF) << 16) | ((K >> 16) & 0xFFFF);
        g1[3] = (32 << 16) | ((M >> 16) & 0xFFFF);    // tile_dim0=32 | dim1 hi
        g1[4] = 64;                                   // tile_dim1=64
        g1[5] = K;                                    // tensor_dim0_stride lo32
        g1[6] = 0;
        g1[7] = 0;
        i32x4 z4 = {0, 0, 0, 0};
#ifdef CTI_TDM_6ARG
        i32x8 z8 = {0, 0, 0, 0, 0, 0, 0, 0};
        __builtin_amdgcn_tensor_load_to_lds(g0, g1, z4, z4, z8, 0);
#else
        __builtin_amdgcn_tensor_load_to_lds(g0, g1, z4, z4, 0);
#endif
        __asm volatile("" ::: "memory");  // LDS written behind compiler's back
    };
    if (wid == 0) tdm_issue(0, 0);        // prime the pipeline
#endif

    for (int s = 0; s < nsteps; ++s) {
        const int k0  = s << 5;
        const int buf = s & 1;
#ifdef CTI_USE_TDM
        if (wid == 0 && s + 1 < nsteps) tdm_issue(buf ^ 1, k0 + 32);
#else
        {   // cooperative vector load of raw A tile
            const int r = tid >> 2, c = (tid & 3) * 8;
            const float* ap = A + (size_t)(m0 + r) * K + k0 + c;
            float4 a0 = *(const float4*)ap;
            float4 a1 = *(const float4*)(ap + 4);
            *(float4*)&Araw[buf][r][c]     = a0;
            *(float4*)&Araw[buf][r][c + 4] = a1;
        }
#endif
        // ---- stage W tile (32k x 64n f32 -> transposed bf16 hi/lo) ----
        {
            float4 b40 = {0.f, 0.f, 0.f, 0.f};
            float4 b41 = {0.f, 0.f, 0.f, 0.f};
            const float* wp = W + (size_t)(k0 + bk) * N + n0 + bn;
            if (n0 + bn < N)     b40 = *(const float4*)wp;
            if (n0 + bn + 4 < N) b41 = *(const float4*)(wp + 4);
            float bv[8] = {b40.x, b40.y, b40.z, b40.w, b41.x, b41.y, b41.z, b41.w};
#pragma unroll
            for (int j = 0; j < 8; ++j) {
                __bf16 h = (__bf16)bv[j];
                Bhi[bn + j][bk] = h;
                Blo[bn + j][bk] = (__bf16)(bv[j] - (float)h);
            }
            if (k0 + 32 < K)   // global_prefetch_b8 of next W tile
                __builtin_prefetch(W + (size_t)(k0 + 32 + bk) * N + n0 + bn, 0, 1);
        }
#ifdef CTI_USE_TDM
        if (wid == 0) {
            if (s + 1 < nsteps) {
                __builtin_amdgcn_s_wait_tensorcnt(1);  // current tile done, next in flight
            } else {
                __builtin_amdgcn_s_wait_tensorcnt(0);  // drain
            }
        }
#endif
        __syncthreads();

        // ---- A fragment: raw f32 from LDS, split hi/lo on read ----
        // lanes 0-15 K {0..15}; lanes 16-31 K {16..31} (matches B)
        const int arow = wm * 16 + (lane & 15);
        const int akb  = (lane < 16) ? 0 : 16;
        const float* arp = &Araw[buf][arow][akb];
        float4 c0 = *(const float4*)(arp);
        float4 c1 = *(const float4*)(arp + 4);
        float4 c2 = *(const float4*)(arp + 8);
        float4 c3 = *(const float4*)(arp + 12);
        float va[16] = {c0.x, c0.y, c0.z, c0.w, c1.x, c1.y, c1.z, c1.w,
                        c2.x, c2.y, c2.z, c2.w, c3.x, c3.y, c3.z, c3.w};
        v16bf afh, afl;
#pragma unroll
        for (int j = 0; j < 16; ++j) {
            __bf16 h = (__bf16)va[j];
            afh[j] = h;
            afl[j] = (__bf16)(va[j] - (float)h);
        }

        // ---- two B fragments (16 contiguous K per lane) ----
        const int bc0 = wn * 32 + (lane & 15);
        const int bkb = (lane < 16) ? 0 : 16;
        v16bf bfh0 = *(const v16bf*)&Bhi[bc0][bkb];
        v16bf bfl0 = *(const v16bf*)&Blo[bc0][bkb];
        v16bf bfh1 = *(const v16bf*)&Bhi[bc0 + 16][bkb];
        v16bf bfl1 = *(const v16bf*)&Blo[bc0 + 16][bkb];

        acc0 = __builtin_amdgcn_wmma_f32_16x16x32_bf16(false, afh, false, bfh0,
                                                       (short)0, acc0, false, false);
        acc0 = __builtin_amdgcn_wmma_f32_16x16x32_bf16(false, afl, false, bfh0,
                                                       (short)0, acc0, false, false);
        acc0 = __builtin_amdgcn_wmma_f32_16x16x32_bf16(false, afh, false, bfl0,
                                                       (short)0, acc0, false, false);
        acc1 = __builtin_amdgcn_wmma_f32_16x16x32_bf16(false, afh, false, bfh1,
                                                       (short)0, acc1, false, false);
        acc1 = __builtin_amdgcn_wmma_f32_16x16x32_bf16(false, afl, false, bfh1,
                                                       (short)0, acc1, false, false);
        acc1 = __builtin_amdgcn_wmma_f32_16x16x32_bf16(false, afh, false, bfl1,
                                                       (short)0, acc1, false, false);
        __syncthreads();
    }

    // ---- epilogue: VGPR r holds M=r (lanes 0-15) / M=r+8 (lanes 16-31) ----
    const int mbase = m0 + wm * 16 + ((lane < 16) ? 0 : 8);
    const int nn0   = n0 + wn * 32 + (lane & 15);
    const int nn1   = nn0 + 16;
    if (nn0 < N) {
        const float bb = bias ? bias[nn0] : 0.f;
#pragma unroll
        for (int r2 = 0; r2 < 8; ++r2) {
            const size_t off = (size_t)(mbase + r2) * N + nn0;
            float v = acc0[r2] + bb;
            if (R) v += R[off];
            C[off] = v;
        }
    }
    if (nn1 < N) {
        const float bb = bias ? bias[nn1] : 0.f;
#pragma unroll
        for (int r2 = 0; r2 < 8; ++r2) {
            const size_t off = (size_t)(mbase + r2) * N + nn1;
            float v = acc1[r2] + bb;
            if (R) v += R[off];
            C[off] = v;
        }
    }
}

// =====================================================================
// Kernel 4: fused MSDA sampling: softmax(12) + 3-level 4-point bilinear
// gather + weighted sum.  One wave per (token, head).
// =====================================================================
__global__ __launch_bounds__(256) void k_msda(
    const float* __restrict__ value, const float* __restrict__ soff,
    const float* __restrict__ awl, float* __restrict__ attn)
{
    const int w = blockIdx.x * 8 + (threadIdx.x >> 5);
    const int lane = threadIdx.x & 31;
    if (w >= NTOK * NHEAD) return;
    const int i = w / NHEAD;
    const int h = w - i * NHEAD;

    const int lstart[3] = {0, 4096, 5120};
    const int lH[3] = {64, 32, 16};
    const int lW[3] = {64, 32, 16};

    int lvl, idx;
    if (i < 4096)      { lvl = 0; idx = i; }
    else if (i < 5120) { lvl = 1; idx = i - 4096; }
    else               { lvl = 2; idx = i - 5120; }
    const int py = idx / lW[lvl];
    const int px = idx - py * lW[lvl];
    const float rx = (px + 0.5f) / (float)lW[lvl];
    const float ry = (py + 0.5f) / (float)lH[lvl];

    float lg[12];
    const float* ap = awl + (size_t)i * 72 + h * 12;
    float mx = -1e30f;
#pragma unroll
    for (int j = 0; j < 12; ++j) { lg[j] = ap[j]; mx = fmaxf(mx, lg[j]); }
    float s = 0.f;
#pragma unroll
    for (int j = 0; j < 12; ++j) { lg[j] = expf(lg[j] - mx); s += lg[j]; }
    const float inv = 1.f / s;

    const float* op = soff + (size_t)i * 144 + h * 24;
    float accx = 0.f, accy = 0.f;

#pragma unroll
    for (int l = 0; l < 3; ++l) {
        const int Hl = lH[l], Wl = lW[l], st = lstart[l];
#pragma unroll
        for (int p = 0; p < 4; ++p) {
            const float ox = op[(l * 4 + p) * 2 + 0];
            const float oy = op[(l * 4 + p) * 2 + 1];
            const float lx = rx + ox / (float)Wl;
            const float ly = ry + oy / (float)Hl;
            const float xx = lx * Wl - 0.5f;
            const float yy = ly * Hl - 0.5f;
            const float x0f = floorf(xx), y0f = floorf(yy);
            const float wx = xx - x0f, wy = yy - y0f;
            const int x0 = (int)x0f, y0 = (int)y0f;
            const float aw = lg[l * 4 + p] * inv;
            const float tw[4] = { aw * (1.f - wx) * (1.f - wy),
                                  aw * wx * (1.f - wy),
                                  aw * (1.f - wx) * wy,
                                  aw * wx * wy };
            const int tx[4] = { x0, x0 + 1, x0, x0 + 1 };
            const int ty[4] = { y0, y0, y0 + 1, y0 + 1 };
#pragma unroll
            for (int t = 0; t < 4; ++t) {
                if (tx[t] >= 0 && tx[t] < Wl && ty[t] >= 0 && ty[t] < Hl) {
                    const size_t row = (size_t)(st + ty[t] * Wl + tx[t]);
                    const float2 v = *(const float2*)(value + row * DIMC + h * HDIM + lane * 2);
                    accx += tw[t] * v.x;
                    accy += tw[t] * v.y;
                }
            }
        }
    }
    float2 o; o.x = accx; o.y = accy;
    *(float2*)(attn + (size_t)i * DIMC + h * HDIM + lane * 2) = o;
}

// =====================================================================
// Kernel 5: depthwise 3x3 ('SAME', zero pad) per pyramid level + exact GELU
// =====================================================================
__global__ __launch_bounds__(256) void k_dwgelu(
    const float* __restrict__ x, const float* __restrict__ dw,
    const float* __restrict__ db, float* __restrict__ y)
{
    const int t = blockIdx.x * 256 + threadIdx.x;
    if (t >= NTOK * 24) return;
    const int tok = t / 24;
    const int c   = (t - tok * 24) * 4;

    int base, hh, ww, pos;
    if (tok < 4096)      { base = 0;    hh = 64; ww = 64; pos = tok; }
    else if (tok < 5120) { base = 4096; hh = 32; ww = 32; pos = tok - 4096; }
    else                 { base = 5120; hh = 16; ww = 16; pos = tok - 5120; }
    const int yy = pos / ww;
    const int xx = pos - yy * ww;

    float4 acc = *(const float4*)(db + c);
#pragma unroll
    for (int dy = -1; dy <= 1; ++dy) {
#pragma unroll
        for (int dx = -1; dx <= 1; ++dx) {
            const int y2 = yy + dy, x2 = xx + dx;
            if (y2 < 0 || y2 >= hh || x2 < 0 || x2 >= ww) continue;
            const float4 v = *(const float4*)(x + (size_t)(base + y2 * ww + x2) * 96 + c);
            const int k = (dy + 1) * 3 + (dx + 1);
            acc.x += v.x * dw[(c + 0) * 9 + k];
            acc.y += v.y * dw[(c + 1) * 9 + k];
            acc.z += v.z * dw[(c + 2) * 9 + k];
            acc.w += v.w * dw[(c + 3) * 9 + k];
        }
    }
    const float is2 = 0.70710678118654752f;
    acc.x = 0.5f * acc.x * (1.f + erff(acc.x * is2));
    acc.y = 0.5f * acc.y * (1.f + erff(acc.y * is2));
    acc.z = 0.5f * acc.z * (1.f + erff(acc.z * is2));
    acc.w = 0.5f * acc.w * (1.f + erff(acc.w * is2));
    *(float4*)(y + (size_t)tok * 96 + c) = acc;
}

// =====================================================================
// Launch
// =====================================================================
extern "C" void kernel_launch(void* const* d_in, const int* in_sizes, int n_in,
                              void* d_out, int out_size, void* d_ws, size_t ws_size,
                              hipStream_t stream) {
    const float* query = (const float*)d_in[0];
    const float* feat  = (const float*)d_in[1];
    const float* qn_g  = (const float*)d_in[7];
    const float* qn_b  = (const float*)d_in[8];
    const float* fn_g  = (const float*)d_in[9];
    const float* fn_b  = (const float*)d_in[10];
    const float* iqn_g = (const float*)d_in[11];
    const float* iqn_b = (const float*)d_in[12];
    const float* ifn_g = (const float*)d_in[13];
    const float* ifn_b = (const float*)d_in[14];
    const float* ffnn_g = (const float*)d_in[15];
    const float* ffnn_b = (const float*)d_in[16];
    const float* so_W  = (const float*)d_in[17];
    const float* so_b  = (const float*)d_in[18];
    const float* aw_W  = (const float*)d_in[19];
    const float* aw_b  = (const float*)d_in[20];
    const float* vp_W  = (const float*)d_in[21];
    const float* vp_b  = (const float*)d_in[22];
    const float* op_W  = (const float*)d_in[23];
    const float* op_b  = (const float*)d_in[24];
    const float* fc1_W = (const float*)d_in[25];
    const float* fc1_b = (const float*)d_in[26];
    const float* dw_W  = (const float*)d_in[27];
    const float* dw_b  = (const float*)d_in[28];
    const float* fc2_W = (const float*)d_in[29];
    const float* fc2_b = (const float*)d_in[30];
    float* out = (float*)d_out;

    float* ws = (float*)d_ws;
    const size_t ND = (size_t)NTOK * DIMC;
    float* qn    = ws;
    float* iqn   = ws + 1 * ND;
    float* ifn   = ws + 2 * ND;
    float* value = ws + 3 * ND;
    float* attn  = ws + 4 * ND;
    float* out1  = ws + 5 * ND;
    float* ffn   = ws + 6 * ND;
    float* soffb = ws + 7 * ND;
    float* awlg  = soffb + (size_t)NTOK * 144;
    float* fc1o  = awlg + (size_t)NTOK * 72;
    float* gel   = fc1o + (size_t)NTOK * 96;

    k_ln_fuse<<<672, 256, 0, stream>>>(query, feat, qn_g, qn_b, fn_g, fn_b,
                                       iqn_g, iqn_b, ifn_g, ifn_b, qn, iqn, ifn);
    k_gemm<<<dim3(84, 6), 256, 0, stream>>>(ifn, vp_W, vp_b, nullptr, value, NTOK, 384, 384);
    k_gemm<<<dim3(84, 3), 256, 0, stream>>>(iqn, so_W, so_b, nullptr, soffb, NTOK, 144, 384);
    k_gemm<<<dim3(84, 2), 256, 0, stream>>>(iqn, aw_W, aw_b, nullptr, awlg, NTOK, 72, 384);
    k_msda<<<4032, 256, 0, stream>>>(value, soffb, awlg, attn);
    k_gemm<<<dim3(84, 6), 256, 0, stream>>>(attn, op_W, op_b, qn, out1, NTOK, 384, 384);
    k_ln<<<672, 256, 0, stream>>>(out1, ffnn_g, ffnn_b, ffn);
    k_gemm<<<dim3(84, 2), 256, 0, stream>>>(ffn, fc1_W, fc1_b, nullptr, fc1o, NTOK, 96, 384);
    k_dwgelu<<<504, 256, 0, stream>>>(fc1o, dw_W, dw_b, gel);
    k_gemm<<<dim3(84, 6), 256, 0, stream>>>(gel, fc2_W, fc2_b, out1, out, NTOK, 384, 96);
}
